// GNN_41549513621974
// MI455X (gfx1250) — compile-verified, matrix operand order checked by reading, not run on hardware
//
#include <hip/hip_runtime.h>
#include <math.h>

// Problem constants (match reference)
constexpr int N    = 4096;
constexpr int E    = 131072;
constexpr int F_IN = 512;
constexpr int H    = 64;
constexpr int C    = 16;
constexpr int Q    = 6;

typedef __attribute__((ext_vector_type(16))) __bf16 v16bf;
typedef __attribute__((ext_vector_type(8)))  float  v8f;

union FragBF { v16bf v; uint4 u[2]; };

// ---------------- generic helpers ----------------
__global__ void k_fill(float* __restrict__ p, float val, size_t n) {
  size_t i = (size_t)blockIdx.x * blockDim.x + threadIdx.x;
  size_t stride = (size_t)gridDim.x * blockDim.x;
  for (; i < n; i += stride) p[i] = val;
}

__global__ void k_zero_small(float* p, int n) {
  int t = threadIdx.x;
  if (t < n) p[t] = 0.0f;
}

// ---------------- edge gates ----------------
__global__ __launch_bounds__(64) void k_edge_gate1(
    const float* __restrict__ x, const int* __restrict__ src, const int* __restrict__ dst,
    const float* __restrict__ W1, const float* __restrict__ b1,
    const float* __restrict__ W2, const float* __restrict__ b2,
    float* __restrict__ s_out) {
  __shared__ float d[F_IN];
  __shared__ float red[64];
  int e = blockIdx.x, t = threadIdx.x;
  int is = src[e], id = dst[e];
  for (int f = t; f < F_IN; f += 64)
    d[f] = x[(size_t)is * F_IN + f] - x[(size_t)id * F_IN + f];
  __syncthreads();
  float acc = b1[t];
  for (int f = 0; f < F_IN; ++f) acc = fmaf(d[f], W1[f * H + t], acc);
  acc = fmaxf(acc, 0.0f);
  red[t] = acc * W2[t];
  __syncthreads();
  for (int k = 32; k > 0; k >>= 1) { if (t < k) red[t] += red[t + k]; __syncthreads(); }
  if (t == 0) {
    float agg = red[0] + b2[0];
    agg *= agg;
    float s = 1.0f / (1.0f + __expf(-agg));
    s_out[e] = fminf(fmaxf(s, 0.0f), 1.0f);
  }
}

__global__ __launch_bounds__(32) void k_edge_gate2(
    const float* __restrict__ x1, const int* __restrict__ src, const int* __restrict__ dst,
    const float* __restrict__ W1, const float* __restrict__ b1,
    const float* __restrict__ W2, const float* __restrict__ b2,
    float* __restrict__ s_out) {
  __shared__ float d[H];
  __shared__ float red[32];
  int e = blockIdx.x, t = threadIdx.x;
  int is = src[e], id = dst[e];
  for (int f = t; f < H; f += 32)
    d[f] = x1[(size_t)is * H + f] - x1[(size_t)id * H + f];
  __syncthreads();
  float acc = b1[t];
  for (int f = 0; f < H; ++f) acc = fmaf(d[f], W1[f * 32 + t], acc);
  acc = fmaxf(acc, 0.0f);
  red[t] = acc * W2[t];
  __syncthreads();
  for (int k = 16; k > 0; k >>= 1) { if (t < k) red[t] += red[t + k]; __syncthreads(); }
  if (t == 0) {
    float agg = red[0] + b2[0];
    agg *= agg;
    float s = 1.0f / (1.0f + __expf(-agg));
    s_out[e] = fminf(fmaxf(s, 0.0f), 1.0f);
  }
}

// ---------------- small dense GEMM (x@W), N*Hc threads ----------------
__global__ void k_gemm_naive(const float* __restrict__ X, const float* __restrict__ W,
                             float* __restrict__ Y, int F, int Hc) {
  int idx = blockIdx.x * blockDim.x + threadIdx.x;
  if (idx >= N * Hc) return;
  int i = idx / Hc, j = idx % Hc;
  const float* xr = X + (size_t)i * F;
  float acc = 0.0f;
  for (int f = 0; f < F; ++f) acc = fmaf(xr[f], W[f * Hc + j], acc);
  Y[idx] = acc;
}

// ---------------- GCN pieces ----------------
__global__ void k_deg_edges(const int* __restrict__ dst, const float* __restrict__ ew,
                            float* __restrict__ deg) {
  int e = blockIdx.x * blockDim.x + threadIdx.x;
  if (e < E) atomicAdd(&deg[dst[e]], ew[e]);
}

__global__ void k_dinv(const float* __restrict__ deg, float* __restrict__ dis) {
  int i = blockIdx.x * blockDim.x + threadIdx.x;
  if (i < N) {
    float d = deg[i];
    dis[i] = (d > 0.0f) ? rsqrtf(fmaxf(d, 1e-12f)) : 0.0f;
  }
}

__global__ void k_gcn_init(const float* __restrict__ h, const float* __restrict__ dis,
                           const float* __restrict__ b, float* __restrict__ out, int Hc) {
  int idx = blockIdx.x * blockDim.x + threadIdx.x;
  if (idx >= N * Hc) return;
  int i = idx / Hc, j = idx % Hc;
  out[idx] = b[j] + dis[i] * dis[i] * h[idx];   // self-loop term, weight 1
}

__global__ void k_gcn_edges(const int* __restrict__ src, const int* __restrict__ dst,
                            const float* __restrict__ ew, const float* __restrict__ dis,
                            const float* __restrict__ h, float* __restrict__ out, int Hc) {
  long long idx = (long long)blockIdx.x * blockDim.x + threadIdx.x;
  if (idx >= (long long)E * Hc) return;
  int e = (int)(idx / Hc), j = (int)(idx % Hc);
  int s = src[e], d = dst[e];
  float v = dis[s] * ew[e] * dis[d] * h[(size_t)s * Hc + j];
  atomicAdd(&out[(size_t)d * Hc + j], v);
}

__global__ void k_relu(float* __restrict__ p, int n) {
  int i = blockIdx.x * blockDim.x + threadIdx.x;
  if (i < n) p[i] = fmaxf(p[i], 0.0f);
}

// ---------------- dense adjacency ----------------
__global__ void k_scatter(const int* __restrict__ src, const int* __restrict__ dst,
                          const float* __restrict__ s, float* __restrict__ A) {
  int e = blockIdx.x * blockDim.x + threadIdx.x;
  if (e < E) atomicAdd(&A[(size_t)src[e] * N + dst[e]], s[e]);
}

__global__ void k_cvt_bf16(const float* __restrict__ A, unsigned short* __restrict__ Abf, size_t n) {
  size_t i = (size_t)blockIdx.x * blockDim.x + threadIdx.x;
  size_t stride = (size_t)gridDim.x * blockDim.x;
  for (; i < n; i += stride) {
    union { float f; unsigned int u; } c; c.f = A[i];
    unsigned int u = c.u;
    Abf[i] = (unsigned short)((u + 0x7FFFu + ((u >> 16) & 1u)) >> 16);  // RNE f32->bf16
  }
}

// ---------------- WMMA GEMM: aa = A(bf16) @ A(bf16)^T, f32 accumulate ----------------
// One wave computes a 16x64 output tile (A-fragment reused x4). The full 32 MB bf16
// matrix is L2-resident (192 MB L2), so global fragment loads hit L2 after warm-up.
__global__ __launch_bounds__(256) void k_wmma_aat(const unsigned short* __restrict__ Abf,
                                                  float* __restrict__ aa) {
  const int lane = threadIdx.x & 31;
  const int wave = threadIdx.x >> 5;
  const int m0  = (blockIdx.y * 8 + wave) * 16;   // 16-row tile
  const int n00 = blockIdx.x * 64;                // 64-col tile
  const int r  = lane & 15;
  const int hi = lane >> 4;

  v8f acc[4] = {};
  const unsigned short* rowA = Abf + (size_t)(m0 + r) * N;
  const unsigned short* rowB0 = Abf + (size_t)(n00 +  0 + r) * N;
  const unsigned short* rowB1 = Abf + (size_t)(n00 + 16 + r) * N;
  const unsigned short* rowB2 = Abf + (size_t)(n00 + 32 + r) * N;
  const unsigned short* rowB3 = Abf + (size_t)(n00 + 48 + r) * N;
  const unsigned short* rowB[4] = {rowB0, rowB1, rowB2, rowB3};

  for (int k0 = 0; k0 < N; k0 += 32) {
    FragBF fa;
    // 16-bit A fragment layout: lanes 0-15 hold K=0..7,16..23; lanes 16-31 hold K=8..15,24..31
    fa.u[0] = *(const uint4*)(rowA + k0 + hi * 8);
    fa.u[1] = *(const uint4*)(rowA + k0 + 16 + hi * 8);
    __builtin_prefetch(rowA + k0 + 128, 0, 1);    // global_prefetch next k-tiles
#pragma unroll
    for (int t = 0; t < 4; ++t) {
      FragBF fb;   // B = A^T: lane n reads row (n00+t*16+n) of Abf, symmetric packing
      fb.u[0] = *(const uint4*)(rowB[t] + k0 + hi * 8);
      fb.u[1] = *(const uint4*)(rowB[t] + k0 + 16 + hi * 8);
      acc[t] = __builtin_amdgcn_wmma_f32_16x16x32_bf16(
          false, fa.v, false, fb.v, (short)0, acc[t], false, false);
    }
  }
#pragma unroll
  for (int t = 0; t < 4; ++t) {
#pragma unroll
    for (int j = 0; j < 8; ++j) {
      // 32-bit C/D layout: lanes 0-15 -> M=j, lanes 16-31 -> M=8+j; N=lane&15
      int row = m0 + hi * 8 + j;
      int col = n00 + t * 16 + r;
      aa[(size_t)row * N + col] = acc[t][j];
    }
  }
}

// ---------------- SVD helper kernels ----------------
__global__ __launch_bounds__(256) void k_matmat_AX(const float* __restrict__ A,
                                                   const float* __restrict__ X,
                                                   float* __restrict__ Y) {
  __shared__ float red[256 * Q];
  int i = blockIdx.x, t = threadIdx.x;
  const float* row = A + (size_t)i * N;
  float acc[Q];
#pragma unroll
  for (int q = 0; q < Q; ++q) acc[q] = 0.0f;
  for (int k = t; k < N; k += 256) {
    float a = row[k];
#pragma unroll
    for (int q = 0; q < Q; ++q) acc[q] = fmaf(a, X[k * Q + q], acc[q]);
  }
#pragma unroll
  for (int q = 0; q < Q; ++q) red[t * Q + q] = acc[q];
  __syncthreads();
  for (int s = 128; s > 0; s >>= 1) {
    if (t < s)
#pragma unroll
      for (int q = 0; q < Q; ++q) red[t * Q + q] += red[(t + s) * Q + q];
    __syncthreads();
  }
  if (t < Q) Y[i * Q + t] = red[t];
}

__global__ __launch_bounds__(256) void k_matmat_ATX(const float* __restrict__ A,
                                                    const float* __restrict__ X,
                                                    float* __restrict__ Z) {
  int k = blockIdx.x * 256 + threadIdx.x;   // column index
  float acc[Q];
#pragma unroll
  for (int q = 0; q < Q; ++q) acc[q] = 0.0f;
  for (int i = 0; i < N; ++i) {
    float a = A[(size_t)i * N + k];
#pragma unroll
    for (int q = 0; q < Q; ++q) acc[q] = fmaf(a, X[i * Q + q], acc[q]);
  }
#pragma unroll
  for (int q = 0; q < Q; ++q) Z[k * Q + q] = acc[q];
}

// In-place modified Gram-Schmidt QR of M [N x Q]; one block.
__global__ __launch_bounds__(256) void k_qr6(float* __restrict__ M) {
  __shared__ float red[256];
  int t = threadIdx.x;
  for (int j = 0; j < Q; ++j) {
    float s = 0.0f;
    for (int i = t; i < N; i += 256) { float v = M[i * Q + j]; s += v * v; }
    red[t] = s; __syncthreads();
    for (int k = 128; k > 0; k >>= 1) { if (t < k) red[t] += red[t + k]; __syncthreads(); }
    float inv = rsqrtf(fmaxf(red[0], 1e-30f)); __syncthreads();
    for (int i = t; i < N; i += 256) M[i * Q + j] *= inv;
    __syncthreads();
    for (int l = j + 1; l < Q; ++l) {
      float d = 0.0f;
      for (int i = t; i < N; i += 256) d += M[i * Q + j] * M[i * Q + l];
      red[t] = d; __syncthreads();
      for (int k = 128; k > 0; k >>= 1) { if (t < k) red[t] += red[t + k]; __syncthreads(); }
      float dot = red[0]; __syncthreads();
      for (int i = t; i < N; i += 256) M[i * Q + l] -= dot * M[i * Q + j];
      __syncthreads();
    }
  }
}

__global__ __launch_bounds__(256) void k_gram(const float* __restrict__ Bt, float* __restrict__ G) {
  __shared__ float red[256];
  int a = blockIdx.x / Q, b = blockIdx.x % Q, t = threadIdx.x;
  float acc = 0.0f;
  for (int n = t; n < N; n += 256) acc = fmaf(Bt[n * Q + a], Bt[n * Q + b], acc);
  red[t] = acc; __syncthreads();
  for (int s = 128; s > 0; s >>= 1) { if (t < s) red[t] += red[t + s]; __syncthreads(); }
  if (t == 0) G[blockIdx.x] = red[0];
}

// 6x6 symmetric Jacobi eigensolver, single thread (sorted descending).
__global__ void k_jacobi6(const float* __restrict__ G, float* __restrict__ U, float* __restrict__ eig) {
  if (threadIdx.x != 0 || blockIdx.x != 0) return;
  float a[Q][Q], v[Q][Q];
  for (int i = 0; i < Q; ++i)
    for (int j = 0; j < Q; ++j) { a[i][j] = G[i * Q + j]; v[i][j] = (i == j) ? 1.0f : 0.0f; }
  for (int sweep = 0; sweep < 30; ++sweep) {
    for (int p = 0; p < Q - 1; ++p)
      for (int q = p + 1; q < Q; ++q) {
        float apq = a[p][q];
        if (fabsf(apq) < 1e-14f) continue;
        float theta = 0.5f * (a[q][q] - a[p][p]) / apq;
        float tt = copysignf(1.0f, theta) / (fabsf(theta) + sqrtf(theta * theta + 1.0f));
        float c = rsqrtf(tt * tt + 1.0f), s = tt * c;
        for (int k = 0; k < Q; ++k) {
          float akp = a[k][p], akq = a[k][q];
          a[k][p] = c * akp - s * akq; a[k][q] = s * akp + c * akq;
        }
        for (int k = 0; k < Q; ++k) {
          float apk = a[p][k], aqk = a[q][k];
          a[p][k] = c * apk - s * aqk; a[q][k] = s * apk + c * aqk;
        }
        for (int k = 0; k < Q; ++k) {
          float vkp = v[k][p], vkq = v[k][q];
          v[k][p] = c * vkp - s * vkq; v[k][q] = s * vkp + c * vkq;
        }
      }
  }
  int idx[Q];
  for (int q = 0; q < Q; ++q) idx[q] = q;
  for (int i = 0; i < Q - 1; ++i)
    for (int j = 0; j < Q - 1 - i; ++j)
      if (a[idx[j]][idx[j]] < a[idx[j + 1]][idx[j + 1]]) { int tmp = idx[j]; idx[j] = idx[j + 1]; idx[j + 1] = tmp; }
  for (int q = 0; q < Q; ++q) {
    eig[q] = a[idx[q]][idx[q]];
    for (int k = 0; k < Q; ++k) U[k * Q + q] = v[k][idx[q]];
  }
}

// V[n,q] = sum_k Bt[n,k] * U[k,q]  (direction only; vi is renormalized downstream)
__global__ void k_makeV(const float* __restrict__ Bt, const float* __restrict__ U,
                        float* __restrict__ V) {
  int idx = blockIdx.x * blockDim.x + threadIdx.x;
  if (idx >= N * Q) return;
  int n = idx / Q, q = idx % Q;
  float acc = 0.0f;
#pragma unroll
  for (int k = 0; k < Q; ++k) acc = fmaf(Bt[n * Q + k], U[k * Q + q], acc);
  V[idx] = acc;
}

// ---------------- nuc loss pieces ----------------
__global__ __launch_bounds__(256) void k_matvec(const float* __restrict__ M,
                                                const float* __restrict__ x, int stride,
                                                float* __restrict__ y) {
  __shared__ float red[256];
  int i = blockIdx.x, t = threadIdx.x;
  const float* row = M + (size_t)i * N;
  float acc = 0.0f;
  for (int k = t; k < N; k += 256) acc = fmaf(row[k], x[k * stride], acc);
  red[t] = acc; __syncthreads();
  for (int s = 128; s > 0; s >>= 1) { if (t < s) red[t] += red[t + s]; __syncthreads(); }
  if (t == 0) y[i] = red[0];
}

__global__ __launch_bounds__(256) void k_dot(const float* __restrict__ a,
                                             const float* __restrict__ b,
                                             float* __restrict__ out) {
  __shared__ float red[256];
  int t = threadIdx.x;
  size_t i = (size_t)blockIdx.x * 256 + t;
  size_t stride = (size_t)gridDim.x * 256;
  float acc = 0.0f;
  for (; i < (size_t)N; i += stride) acc = fmaf(a[i], b[i], acc);
  red[t] = acc; __syncthreads();
  for (int s = 128; s > 0; s >>= 1) { if (t < s) red[t] += red[t + s]; __syncthreads(); }
  if (t == 0) atomicAdd(out, red[0]);
}

__global__ void k_scale_rsqrt(float* __restrict__ v, const float* __restrict__ scal) {
  int i = blockIdx.x * blockDim.x + threadIdx.x;
  if (i < N) v[i] *= rsqrtf(fmaxf(scal[0], 1e-30f));
}

__global__ void k_accum_nuc(float* __restrict__ scal) {
  if (threadIdx.x == 0 && blockIdx.x == 0)
    scal[3] += sqrtf(fabsf(scal[1] / fmaxf(scal[2], 1e-30f)));
}

// elementwise deflation: aa[i,j] *= (1 - v_i v_j)
__global__ void k_deflate_ew(float* __restrict__ aa, const float* __restrict__ v, size_t nn) {
  size_t i = (size_t)blockIdx.x * blockDim.x + threadIdx.x;
  size_t stride = (size_t)gridDim.x * blockDim.x;
  for (; i < nn; i += stride) {
    size_t r = i >> 12, c = i & (N - 1);   // N = 4096 = 2^12
    aa[i] = fmaf(-aa[i], v[r] * v[c], aa[i]);
  }
}

// matmul deflation as rank-1: aa[i,j] -= w_i v_j  (w = aa @ v, pre-deflation)
__global__ void k_deflate_r1(float* __restrict__ aa, const float* __restrict__ w,
                             const float* __restrict__ v, size_t nn) {
  size_t i = (size_t)blockIdx.x * blockDim.x + threadIdx.x;
  size_t stride = (size_t)gridDim.x * blockDim.x;
  for (; i < nn; i += stride) {
    size_t r = i >> 12, c = i & (N - 1);
    aa[i] = fmaf(-w[r], v[c], aa[i]);
  }
}

// ---------------- consistency + outputs ----------------
__global__ __launch_bounds__(256) void k_sum2(const float* __restrict__ s, int n,
                                              float* __restrict__ sum, float* __restrict__ sumsq) {
  __shared__ float r1[256], r2[256];
  int t = threadIdx.x;
  size_t i = (size_t)blockIdx.x * 256 + t;
  size_t stride = (size_t)gridDim.x * 256;
  float a = 0.0f, b = 0.0f;
  for (; i < (size_t)n; i += stride) { float v = s[i]; a += v; b = fmaf(v, v, b); }
  r1[t] = a; r2[t] = b; __syncthreads();
  for (int k = 128; k > 0; k >>= 1) {
    if (t < k) { r1[t] += r1[t + k]; r2[t] += r2[t + k]; }
    __syncthreads();
  }
  if (t == 0) { atomicAdd(sum, r1[0]); atomicAdd(sumsq, r2[0]); }
}

__global__ void k_softmax_out(const float* __restrict__ xout, float* __restrict__ logsm,
                              float* __restrict__ sm, float* __restrict__ calib_acc) {
  int i = blockIdx.x * blockDim.x + threadIdx.x;
  if (i >= N) return;
  float r[C];
  float m = -3.4e38f;
#pragma unroll
  for (int j = 0; j < C; ++j) { r[j] = xout[(size_t)i * C + j]; m = fmaxf(m, r[j]); }
  float se = 0.0f;
#pragma unroll
  for (int j = 0; j < C; ++j) se += __expf(r[j] - m);
  float lse = logf(se) + m;
  float m1 = -3.4e38f, m2 = -3.4e38f;
#pragma unroll
  for (int j = 0; j < C; ++j) {
    float l = r[j] - lse;
    logsm[(size_t)i * C + j] = l;
    sm[(size_t)i * C + j]    = __expf(l);
    if (r[j] > m1) { m2 = m1; m1 = r[j]; }
    else if (r[j] > m2) { m2 = r[j]; }
  }
  atomicAdd(calib_acc, m2 - m1);   // calib_i = -top1 + top2
}

__global__ void k_finalize(const float* __restrict__ scal, float* __restrict__ loss_out,
                           float* __restrict__ calib_out) {
  if (threadIdx.x != 0 || blockIdx.x != 0) return;
  float inv_e = 1.0f / (float)E;
  float m1 = scal[4] * inv_e, c1 = scal[5] * inv_e - m1 * m1;
  float m2 = scal[6] * inv_e, c2 = scal[7] * inv_e - m2 * m2;
  *loss_out  = 0.01f * (c1 + c2) + 0.01f * scal[3];
  *calib_out = scal[8] / (float)N;
}

// =====================================================================
extern "C" void kernel_launch(void* const* d_in, const int* in_sizes, int n_in,
                              void* d_out, int out_size, void* d_ws, size_t ws_size,
                              hipStream_t stream) {
  (void)in_sizes; (void)n_in; (void)out_size; (void)ws_size;
  const float* x    = (const float*)d_in[0];
  const int*   ei   = (const int*)d_in[1];
  const int*   src  = ei;
  const int*   dst  = ei + E;
  const float* g1w  = (const float*)d_in[2];
  const float* g1b  = (const float*)d_in[3];
  const float* g2w  = (const float*)d_in[4];
  const float* g2b  = (const float*)d_in[5];
  const float* p1w1 = (const float*)d_in[6];
  const float* p1b1 = (const float*)d_in[7];
  const float* p1w2 = (const float*)d_in[8];
  const float* p1b2 = (const float*)d_in[9];
  const float* p2w1 = (const float*)d_in[10];
  const float* p2b1 = (const float*)d_in[11];
  const float* p2w2 = (const float*)d_in[12];
  const float* p2b2 = (const float*)d_in[13];
  const float* om1  = (const float*)d_in[14];
  const float* om2  = (const float*)d_in[15];

  const size_t NN = (size_t)N * N;
  const int NC = N * C;

  // output layout: [logsm NC][sm NC][loss 1][s1 E][s2 E][calib 1]
  float* out      = (float*)d_out;
  float* o_logsm  = out;
  float* o_sm     = out + NC;
  float* o_loss   = out + 2 * NC;
  float* o_s1     = out + 2 * NC + 1;
  float* o_s2     = o_s1 + E;
  float* o_calib  = o_s2 + E;

  // workspace layout (~163 MB)
  char* wsb = (char*)d_ws;
  float*          A    = (float*)wsb;                                  // N*N f32
  float*          aa   = A + NN;                                       // N*N f32
  unsigned short* Abf  = (unsigned short*)(aa + NN);                   // N*N bf16
  float*          fb   = (float*)((char*)Abf + NN * sizeof(unsigned short));
  float* h    = fb;            fb += (size_t)N * H;
  float* x1   = fb;            fb += (size_t)N * H;
  float* h2   = fb;            fb += (size_t)N * C;
  float* xout = fb;            fb += (size_t)N * C;
  float* deg  = fb;            fb += N;
  float* dis  = fb;            fb += N;
  float* vi   = fb;            fb += N;
  float* w    = fb;            fb += N;
  float* Qm   = fb;            fb += (size_t)N * Q;
  float* Zm   = fb;            fb += (size_t)N * Q;
  float* Bt   = fb;            fb += (size_t)N * Q;
  float* Vm   = fb;            fb += (size_t)N * Q;
  float* G    = fb;            fb += 36;
  float* U    = fb;            fb += 36;
  float* eig  = fb;            fb += 8;
  float* scal = fb;            // 16 scalars: 0 norm,1 vmv,2 vv,3 nuc,4..7 gate sums,8 calib

  k_zero_small<<<1, 32, 0, stream>>>(scal, 16);

  // ============ graph 1 ============
  k_edge_gate1<<<E, 64, 0, stream>>>(x, src, dst, p1w1, p1b1, p1w2, p1b2, o_s1);

  k_fill<<<64, 256, 0, stream>>>(deg, 1.0f, N);            // self-loop weight 1
  k_deg_edges<<<E / 256, 256, 0, stream>>>(dst, o_s1, deg);
  k_dinv<<<N / 256, 256, 0, stream>>>(deg, dis);

  k_gemm_naive<<<(N * H) / 256, 256, 0, stream>>>(x, g1w, h, F_IN, H);
  k_gcn_init<<<(N * H) / 256, 256, 0, stream>>>(h, dis, g1b, x1, H);
  k_gcn_edges<<<(E * H) / 256, 256, 0, stream>>>(src, dst, o_s1, dis, h, x1, H);
  k_relu<<<(N * H) / 256, 256, 0, stream>>>(x1, N * H);

  k_sum2<<<256, 256, 0, stream>>>(o_s1, E, scal + 4, scal + 5);

  k_fill<<<8192, 256, 0, stream>>>(A, 1e-6f, NN);
  k_scatter<<<E / 256, 256, 0, stream>>>(src, dst, o_s1, A);
  k_cvt_bf16<<<8192, 256, 0, stream>>>(A, Abf, NN);
  k_wmma_aat<<<dim3(N / 64, N / 128), 256, 0, stream>>>(Abf, aa);

  // randomized SVD range finder (niter = 2)
  k_matmat_AX<<<N, 256, 0, stream>>>(A, om1, Qm);
  k_qr6<<<1, 256, 0, stream>>>(Qm);
  for (int it = 0; it < 2; ++it) {
    k_matmat_ATX<<<N / 256, 256, 0, stream>>>(A, Qm, Zm);
    k_qr6<<<1, 256, 0, stream>>>(Zm);
    k_matmat_AX<<<N, 256, 0, stream>>>(A, Zm, Qm);
    k_qr6<<<1, 256, 0, stream>>>(Qm);
  }
  k_matmat_ATX<<<N / 256, 256, 0, stream>>>(A, Qm, Bt);   // Bt = A^T Q  (= B^T)
  k_gram<<<Q * Q, 256, 0, stream>>>(Bt, G);
  k_jacobi6<<<1, 1, 0, stream>>>(G, U, eig);
  k_makeV<<<(N * Q + 255) / 256, 256, 0, stream>>>(Bt, U, Vm);

  // nuc loss, elementwise deflation branch (deflate_matmul=False)
  for (int k = 0; k < Q; ++k) {
    k_zero_small<<<1, 32, 0, stream>>>(scal, 3);
    k_matvec<<<N, 256, 0, stream>>>(aa, Vm + k, Q, vi);
    k_dot<<<64, 256, 0, stream>>>(vi, vi, scal + 0);
    k_scale_rsqrt<<<N / 256, 256, 0, stream>>>(vi, scal);
    k_matvec<<<N, 256, 0, stream>>>(aa, vi, 1, w);
    k_dot<<<64, 256, 0, stream>>>(vi, w, scal + 1);
    k_dot<<<64, 256, 0, stream>>>(vi, vi, scal + 2);
    k_accum_nuc<<<1, 1, 0, stream>>>(scal);
    if (k > 1 && k < Q - 1)   // k = Q-1 deflation result is never read
      k_deflate_ew<<<8192, 256, 0, stream>>>(aa, vi, NN);
  }

  // ============ graph 2 ============
  k_edge_gate2<<<E, 32, 0, stream>>>(x1, src, dst, p2w1, p2b1, p2w2, p2b2, o_s2);

  k_fill<<<64, 256, 0, stream>>>(deg, 1.0f, N);
  k_deg_edges<<<E / 256, 256, 0, stream>>>(dst, o_s2, deg);
  k_dinv<<<N / 256, 256, 0, stream>>>(deg, dis);

  k_gemm_naive<<<(N * C) / 256, 256, 0, stream>>>(x1, g2w, h2, H, C);
  k_gcn_init<<<(N * C) / 256, 256, 0, stream>>>(h2, dis, g2b, xout, C);
  k_gcn_edges<<<(E * C) / 256, 256, 0, stream>>>(src, dst, o_s2, dis, h2, xout, C);

  k_sum2<<<256, 256, 0, stream>>>(o_s2, E, scal + 6, scal + 7);

  k_fill<<<8192, 256, 0, stream>>>(A, 1e-6f, NN);
  k_scatter<<<E / 256, 256, 0, stream>>>(src, dst, o_s2, A);
  k_cvt_bf16<<<8192, 256, 0, stream>>>(A, Abf, NN);
  k_wmma_aat<<<dim3(N / 64, N / 128), 256, 0, stream>>>(Abf, aa);

  k_matmat_AX<<<N, 256, 0, stream>>>(A, om2, Qm);
  k_qr6<<<1, 256, 0, stream>>>(Qm);
  for (int it = 0; it < 2; ++it) {
    k_matmat_ATX<<<N / 256, 256, 0, stream>>>(A, Qm, Zm);
    k_qr6<<<1, 256, 0, stream>>>(Zm);
    k_matmat_AX<<<N, 256, 0, stream>>>(A, Zm, Qm);
    k_qr6<<<1, 256, 0, stream>>>(Qm);
  }
  k_matmat_ATX<<<N / 256, 256, 0, stream>>>(A, Qm, Bt);
  k_gram<<<Q * Q, 256, 0, stream>>>(Bt, G);
  k_jacobi6<<<1, 1, 0, stream>>>(G, U, eig);
  k_makeV<<<(N * Q + 255) / 256, 256, 0, stream>>>(Bt, U, Vm);

  // nuc loss, matmul deflation branch: aa -= (aa v) v^T  (rank-1, w already computed)
  for (int k = 0; k < Q; ++k) {
    k_zero_small<<<1, 32, 0, stream>>>(scal, 3);
    k_matvec<<<N, 256, 0, stream>>>(aa, Vm + k, Q, vi);
    k_dot<<<64, 256, 0, stream>>>(vi, vi, scal + 0);
    k_scale_rsqrt<<<N / 256, 256, 0, stream>>>(vi, scal);
    k_matvec<<<N, 256, 0, stream>>>(aa, vi, 1, w);
    k_dot<<<64, 256, 0, stream>>>(vi, w, scal + 1);
    k_dot<<<64, 256, 0, stream>>>(vi, vi, scal + 2);
    k_accum_nuc<<<1, 1, 0, stream>>>(scal);
    if (k > 1 && k < Q - 1)
      k_deflate_r1<<<8192, 256, 0, stream>>>(aa, w, vi, NN);
  }

  // ============ outputs ============
  k_softmax_out<<<N / 256, 256, 0, stream>>>(xout, o_logsm, o_sm, scal + 8);
  k_finalize<<<1, 1, 0, stream>>>(scal, o_loss, o_calib);
}
